// MultiHeadAttention_4569845203483
// MI455X (gfx1250) — matrix operand order repaired and checked by reading, not compiled, not run on hardware
//
#include <hip/hip_runtime.h>

// Problem constants (match reference)
#define B_  4
#define S_  2048
#define D_  1024
#define H_  16
#define DK_ 64
#define M_  (B_ * S_)   // 8192 rows for all GEMMs

typedef __bf16 bf16;
typedef __attribute__((ext_vector_type(16))) __bf16 v16bf;
typedef __attribute__((ext_vector_type(8)))  __bf16 v8bf;
typedef __attribute__((ext_vector_type(4)))  __bf16 v4bf;
typedef __attribute__((ext_vector_type(8)))  float  v8f;
typedef __attribute__((ext_vector_type(4)))  float  v4f;

#define WMMA_BF16(A, Bm, C) \
  __builtin_amdgcn_wmma_f32_16x16x32_bf16(false, (A), false, (Bm), (short)0, (C), false, false)

// float -> bf16, round-to-nearest-even (avoids relying on __bf16 casts)
__device__ __forceinline__ bf16 f2bf(float f) {
  union { float f; unsigned u; } a; a.f = f;
  unsigned u = a.u + 0x7FFFu + ((a.u >> 16) & 1u);
  union { unsigned short s; bf16 b; } o; o.s = (unsigned short)(u >> 16);
  return o.b;
}

// ---- Fragment loaders: contiguous k = 16*(lane/16) + s, same order for A and B ----

// B-matrix 32x16 bf16: lane n (= lane&15) holds column n; lanes 0-15 K=0..15,
// lanes 16-31 K=16..31 -> 16 contiguous bf16 per lane.
__device__ __forceinline__ v16bf load_b16(const bf16* p) {
  union { v16bf v; v8bf h[2]; } u;
  u.h[0] = *(const v8bf*)(p);
  u.h[1] = *(const v8bf*)(p + 8);
  return u.v;
}

// A-matrix 16x32 bf16: lane<16 holds row (lane&15), K = s..s+15
// with s = 0; lane>=16 same row with s = 16.  rowptr points at (row, k0).
__device__ __forceinline__ v16bf load_a_bf16(const bf16* rowptr, int lane) {
  const int s = (lane >= 16) ? 16 : 0;
  union { v16bf v; v8bf h[2]; } u;
  u.h[0] = *(const v8bf*)(rowptr + s);
  u.h[1] = *(const v8bf*)(rowptr + s + 8);
  return u.v;
}

// Same A fragment but sourced from fp32 memory, converted on the fly.
__device__ __forceinline__ v16bf load_a_f32(const float* rowptr, int lane) {
  const int s = (lane >= 16) ? 16 : 0;
  v4f a = *(const v4f*)(rowptr + s);
  v4f b = *(const v4f*)(rowptr + s + 4);
  v4f c = *(const v4f*)(rowptr + s + 8);
  v4f d = *(const v4f*)(rowptr + s + 12);
  v16bf r;
#pragma unroll
  for (int i = 0; i < 4; ++i) {
    r[i]      = f2bf(a[i]);
    r[4 + i]  = f2bf(b[i]);
    r[8 + i]  = f2bf(c[i]);
    r[12 + i] = f2bf(d[i]);
  }
  return r;
}

// Cross-lane reductions inside each 16-lane half (C-layout rows live per-half)
__device__ __forceinline__ float half_max(float v) {
  v = fmaxf(v, __shfl_xor(v, 1));
  v = fmaxf(v, __shfl_xor(v, 2));
  v = fmaxf(v, __shfl_xor(v, 4));
  v = fmaxf(v, __shfl_xor(v, 8));
  return v;
}
__device__ __forceinline__ float half_sum(float v) {
  v += __shfl_xor(v, 1);
  v += __shfl_xor(v, 2);
  v += __shfl_xor(v, 4);
  v += __shfl_xor(v, 8);
  return v;
}

// ---------------- fp32 -> bf16 conversion (weights) ----------------
__global__ __launch_bounds__(256) void cvt_f32_bf16_kernel(
    const float* __restrict__ src, bf16* __restrict__ dst, int n) {
  int i = (blockIdx.x * blockDim.x + threadIdx.x) * 4;
  if (i < n) {
    v4f x = *(const v4f*)(src + i);
    v4bf y;
#pragma unroll
    for (int c = 0; c < 4; ++c) y[c] = f2bf(x[c]);
    *(v4bf*)(dst + i) = y;
  }
}

// ---------------- WMMA GEMM:  out = (A[M,K] * W^T + bias) * scale --------
// W is bf16 row-major [N,K]; B-matrix column n == W row n (contiguous K).
// Block: 256 threads = 8 waves (4M x 2N), block tile 128M x 64N,
// wave tile 32M x 32N (2x2 frags), K step 32.
// STORE: 0 = bf16 row-major [M,D]; 1 = bf16 V^T [B,H,DK,S]; 2 = fp32 [M,D]
template <bool A_BF16, int STORE>
__global__ __launch_bounds__(256) void gemm_wmma_kernel(
    const float* __restrict__ Xf, const bf16* __restrict__ Xb,
    const bf16* __restrict__ W, const float* __restrict__ bias,
    void* __restrict__ out, float scale) {
  const int lane = threadIdx.x & 31;
  const int wid  = threadIdx.x >> 5;
  const int mw = wid & 3, nw = wid >> 2;
  const int m0 = blockIdx.y * 128 + mw * 32;
  const int n0 = blockIdx.x * 64 + nw * 32;
  const int ln = lane & 15;

  v8f acc[2][2] = {};

  for (int k0 = 0; k0 < D_; k0 += 32) {
    v16bf bfr[2];
#pragma unroll
    for (int nf = 0; nf < 2; ++nf) {
      const int col = n0 + nf * 16 + ln;
      bfr[nf] = load_b16(W + (size_t)col * D_ + k0 + ((lane >= 16) ? 16 : 0));
    }
#pragma unroll
    for (int mf = 0; mf < 2; ++mf) {
      const int row = m0 + mf * 16 + ln;
      v16bf afr;
      if constexpr (A_BF16)
        afr = load_a_bf16(Xb + (size_t)row * D_ + k0, lane);
      else
        afr = load_a_f32(Xf + (size_t)row * D_ + k0, lane);
#pragma unroll
      for (int nf = 0; nf < 2; ++nf)
        acc[mf][nf] = WMMA_BF16(afr, bfr[nf], acc[mf][nf]);
    }
  }

  const int rb = (lane < 16) ? 0 : 8;
#pragma unroll
  for (int mf = 0; mf < 2; ++mf)
#pragma unroll
    for (int nf = 0; nf < 2; ++nf) {
      const int col = n0 + nf * 16 + ln;
      const float bv = bias[col];
#pragma unroll
      for (int j = 0; j < 8; ++j) {
        const int row = m0 + mf * 16 + rb + j;
        const float v = (acc[mf][nf][j] + bv) * scale;
        if constexpr (STORE == 0) {
          ((bf16*)out)[(size_t)row * D_ + col] = f2bf(v);
        } else if constexpr (STORE == 1) {
          const int bb = row >> 11, s = row & (S_ - 1);
          const int hh = col >> 6, d = col & (DK_ - 1);
          ((bf16*)out)[(size_t)((bb * H_ + hh) * DK_ + d) * S_ + s] = f2bf(v);
        } else {
          ((float*)out)[(size_t)row * D_ + col] = v;
        }
      }
    }
}

// ---------------- Flash attention (causal, online softmax) ----------------
// Q pre-scaled by 1/sqrt(DK). Qb,Kb: bf16 [B,S,D]; Vt: bf16 [B,H,DK,S].
// Grid: (S/256, H, B). Block = 256 threads = 8 waves; wave owns 32 queries.
// Key loop step 32; QK^T and PV are both bf16 WMMA; P converts C-layout ->
// A-layout through a per-wave LDS tile (wave-local, DS-ordered, no barrier).
__global__ __launch_bounds__(256) void flash_attn_kernel(
    const bf16* __restrict__ Qb, const bf16* __restrict__ Kb,
    const bf16* __restrict__ Vt, bf16* __restrict__ Ctx) {
  __shared__ bf16 lds_p[8][32 * 32];  // 16 KB: per-wave 32x32 P tile

  const int lane = threadIdx.x & 31;
  const int wid  = threadIdx.x >> 5;
  const int b = blockIdx.z, h = blockIdx.y;
  const int q0 = blockIdx.x * 256 + wid * 32;
  const int ln = lane & 15;
  const int rb = (lane < 16) ? 0 : 8;
  const float NEG = -1.0e30f;

  // Preload Q A-fragments: [mfrag][dstep], d = dstep*32
  v16bf qf[2][2];
#pragma unroll
  for (int mf = 0; mf < 2; ++mf)
#pragma unroll
    for (int ds = 0; ds < 2; ++ds) {
      const int row = q0 + mf * 16 + ln;
      qf[mf][ds] = load_a_bf16(
          Qb + (size_t)(b * S_ + row) * D_ + h * DK_ + ds * 32, lane);
    }

  v8f o[2][4] = {};
  float mi[2][8], li[2][8];
#pragma unroll
  for (int mf = 0; mf < 2; ++mf)
#pragma unroll
    for (int j = 0; j < 8; ++j) { mi[mf][j] = NEG; li[mf][j] = 0.0f; }

  const int kend = q0 + 32;  // causal: key blocks up to & incl. diagonal
  for (int kb = 0; kb < kend; kb += 32) {
    if (kb + 32 < kend) {  // hint next K block toward L2/WGP$
      __builtin_prefetch(Kb + (size_t)(b * S_ + kb + 32 + lane) * D_ + h * DK_, 0, 3);
    }
    // K B-fragments: columns = key indices, contiguous d per lane
    v16bf kf[2][2];
#pragma unroll
    for (int nf = 0; nf < 2; ++nf)
#pragma unroll
      for (int ds = 0; ds < 2; ++ds) {
        const int col = kb + nf * 16 + ln;
        kf[nf][ds] = load_b16(Kb + (size_t)(b * S_ + col) * D_ + h * DK_ +
                              ds * 32 + ((lane >= 16) ? 16 : 0));
      }

    // S = Q K^T (32x32 per wave), fp32 accum
    v8f sc[2][2];
#pragma unroll
    for (int mf = 0; mf < 2; ++mf)
#pragma unroll
      for (int nf = 0; nf < 2; ++nf) {
        v8f a = {};
#pragma unroll
        for (int ds = 0; ds < 2; ++ds) a = WMMA_BF16(qf[mf][ds], kf[nf][ds], a);
        sc[mf][nf] = a;
      }

    // Causal mask (only the diagonal block is partial since kb <= q0)
    if (kb == q0) {
#pragma unroll
      for (int mf = 0; mf < 2; ++mf)
#pragma unroll
        for (int nf = 0; nf < 2; ++nf) {
          const int c = nf * 16 + ln;
#pragma unroll
          for (int j = 0; j < 8; ++j) {
            const int r = mf * 16 + rb + j;
            if (c > r) sc[mf][nf][j] = NEG;
          }
        }
    }

    // Online softmax per m-fragment, then P (bf16) into per-wave LDS tile
#pragma unroll
    for (int mf = 0; mf < 2; ++mf) {
#pragma unroll
      for (int j = 0; j < 8; ++j) {
        float mn = half_max(fmaxf(sc[mf][0][j], sc[mf][1][j]));
        const float mo = mi[mf][j];
        const float m2 = fmaxf(mo, mn);
        const float sf = __expf(mo - m2);
        mi[mf][j] = m2;
        li[mf][j] *= sf;
#pragma unroll
        for (int nf = 0; nf < 4; ++nf) o[mf][nf][j] *= sf;
        const float p0 = __expf(sc[mf][0][j] - m2);
        const float p1 = __expf(sc[mf][1][j] - m2);
        li[mf][j] += half_sum(p0 + p1);
        const int r = mf * 16 + rb + j;
        lds_p[wid][r * 32 + ln]      = f2bf(p0);
        lds_p[wid][r * 32 + 16 + ln] = f2bf(p1);
      }
    }

    // Reload P as A-fragments (wave-local LDS; DS ops are in-order per wave)
    v16bf pf[2];
#pragma unroll
    for (int mf = 0; mf < 2; ++mf)
      pf[mf] = load_a_bf16(&lds_p[wid][(mf * 16 + ln) * 32], lane);

    // V^T B-fragments: columns = head dims d, contiguous keys per lane
    v16bf vf[4];
#pragma unroll
    for (int nf = 0; nf < 4; ++nf) {
      const int d = nf * 16 + ln;
      vf[nf] = load_b16(Vt + (size_t)((b * H_ + h) * DK_ + d) * S_ + kb +
                        ((lane >= 16) ? 16 : 0));
    }

    // O += P V
#pragma unroll
    for (int mf = 0; mf < 2; ++mf)
#pragma unroll
      for (int nf = 0; nf < 4; ++nf)
        o[mf][nf] = WMMA_BF16(pf[mf], vf[nf], o[mf][nf]);
  }

  // Normalize and store context (bf16, [B,S,D])
#pragma unroll
  for (int mf = 0; mf < 2; ++mf)
#pragma unroll
    for (int j = 0; j < 8; ++j) {
      const float inv = 1.0f / li[mf][j];
      const int row = q0 + mf * 16 + rb + j;
#pragma unroll
      for (int nf = 0; nf < 4; ++nf) {
        const int d = nf * 16 + ln;
        Ctx[(size_t)(b * S_ + row) * D_ + h * DK_ + d] =
            f2bf(o[mf][nf][j] * inv);
      }
    }
}

// ------------------------------- launcher ---------------------------------
extern "C" void kernel_launch(void* const* d_in, const int* in_sizes, int n_in,
                              void* d_out, int out_size, void* d_ws,
                              size_t ws_size, hipStream_t stream) {
  (void)in_sizes; (void)n_in; (void)out_size; (void)ws_size;
  const float* query = (const float*)d_in[0];
  const float* key_t = (const float*)d_in[1];
  const float* value = (const float*)d_in[2];
  const float* Wq = (const float*)d_in[3];  const float* bq = (const float*)d_in[4];
  const float* Wk = (const float*)d_in[5];  const float* bk = (const float*)d_in[6];
  const float* Wv = (const float*)d_in[7];  const float* bv = (const float*)d_in[8];
  const float* Wo = (const float*)d_in[9];  const float* bo = (const float*)d_in[10];
  // d_in[11] = causal mask: implemented analytically in-kernel.
  float* out = (float*)d_out;

  // Workspace layout (bf16), 72 MB total
  char* ws = (char*)d_ws;
  const size_t WSZ = (size_t)D_ * D_ * 2;        // 2 MB per weight
  const size_t TSZ = (size_t)M_ * D_ * 2;        // 16 MB per activation
  bf16* Wqb = (bf16*)(ws + 0 * WSZ);
  bf16* Wkb = (bf16*)(ws + 1 * WSZ);
  bf16* Wvb = (bf16*)(ws + 2 * WSZ);
  bf16* Wob = (bf16*)(ws + 3 * WSZ);
  bf16* Qb  = (bf16*)(ws + 4 * WSZ);
  bf16* Kb  = (bf16*)(ws + 4 * WSZ + 1 * TSZ);
  bf16* Vt  = (bf16*)(ws + 4 * WSZ + 2 * TSZ);
  bf16* Ctx = (bf16*)(ws + 4 * WSZ + 3 * TSZ);

  // 1) weights fp32 -> bf16
  {
    const int n = D_ * D_;
    dim3 g((n / 4 + 255) / 256), blk(256);
    cvt_f32_bf16_kernel<<<g, blk, 0, stream>>>(Wq, Wqb, n);
    cvt_f32_bf16_kernel<<<g, blk, 0, stream>>>(Wk, Wkb, n);
    cvt_f32_bf16_kernel<<<g, blk, 0, stream>>>(Wv, Wvb, n);
    cvt_f32_bf16_kernel<<<g, blk, 0, stream>>>(Wo, Wob, n);
  }

  // 2) projections (fp32 A converted on the fly)
  dim3 gg(D_ / 64, M_ / 128), gb(256);
  const float qscale = 0.125f;  // 1/sqrt(DK), folded into Q
  gemm_wmma_kernel<false, 0><<<gg, gb, 0, stream>>>(query, nullptr, Wqb, bq, Qb, qscale);
  gemm_wmma_kernel<false, 0><<<gg, gb, 0, stream>>>(key_t, nullptr, Wkb, bk, Kb, 1.0f);
  gemm_wmma_kernel<false, 1><<<gg, gb, 0, stream>>>(value, nullptr, Wvb, bv, Vt, 1.0f);

  // 3) causal flash attention
  dim3 ga(S_ / 256, H_, B_);
  flash_attn_kernel<<<ga, gb, 0, stream>>>(Qb, Kb, Vt, Ctx);

  // 4) output projection -> fp32
  gemm_wmma_kernel<true, 2><<<gg, gb, 0, stream>>>(nullptr, Ctx, Wob, bo, out, 1.0f);
}
